// NeocortexMemory_5102421148031
// MI455X (gfx1250) — compile-verified, matrix operand order checked by reading, not run on hardware
//
#include <hip/hip_runtime.h>
#include <hip/hip_bf16.h>
#include <math.h>

typedef __bf16 bf16;
typedef __attribute__((ext_vector_type(16))) __bf16 v16bf;
typedef __attribute__((ext_vector_type(8)))  __bf16 v8bf;
typedef __attribute__((ext_vector_type(8)))  float  v8f;

#define NB 65536
#define ND 1024
#define NP 32

// ---------------------------------------------------------------------------
// Conversion kernels
// ---------------------------------------------------------------------------
__global__ __launch_bounds__(256) void f32_to_bf16_kernel(
    const float* __restrict__ src, bf16* __restrict__ dst, int n) {
  int i = blockIdx.x * 256 + threadIdx.x;
  if (i < n) dst[i] = (bf16)src[i];
}

// Pack x (f32 [B,1024]) into first half of combined (bf16 [B,2048])
__global__ __launch_bounds__(256) void pack_x_kernel(
    const float* __restrict__ x, bf16* __restrict__ combined) {
  size_t i = (size_t)blockIdx.x * 256 + threadIdx.x;   // 0..B*D-1
  size_t r = i >> 10;
  int    c = (int)(i & 1023);
  combined[r * 2048 + c] = (bf16)x[i];
}

// l2-normalize the 32 prototype rows
__global__ __launch_bounds__(256) void proto_norm_kernel(
    const float* __restrict__ proto, float* __restrict__ protoN) {
  __shared__ float red[8];
  const int t = threadIdx.x;
  const int row = blockIdx.x;
  float ss = 0.f;
  float v[4];
#pragma unroll
  for (int i = 0; i < 4; i++) {
    int idx = t + 256 * i;
    v[i] = proto[row * ND + idx];
    ss += v[i] * v[i];
  }
  for (int m = 16; m >= 1; m >>= 1) ss += __shfl_xor(ss, m);
  if ((t & 31) == 0) red[t >> 5] = ss;
  __syncthreads();
  float tot = red[0] + red[1] + red[2] + red[3] + red[4] + red[5] + red[6] + red[7];
  float inv = 1.0f / fmaxf(sqrtf(tot), 1e-12f);
#pragma unroll
  for (int i = 0; i < 4; i++) {
    int idx = t + 256 * i;
    protoN[row * ND + idx] = v[i] * inv;
  }
}

// ---------------------------------------------------------------------------
// bf16 WMMA GEMM:  C[M,N] = A[M,K] @ W[N,K]^T + bias[N]   (optional exact gelu)
// Block: 256 threads = 8 waves; block tile 128x128.
// Wave tile 32x64 = 2(M) x 4(N) WMMA fragments -> 8 v_wmma per K-step on
// 12 b128 loads. Grid is N-major (grid.x = N/128 = 8) so 8 consecutive
// blocks share one 128-row A tile -> A streams HBM ~once, re-reads hit L2.
// A fragment layout per CDNA5 ISA (16-bit A 16x32): lane&15 -> M row,
// (lane>>4)*8 -> K base; elems [0..7] at K=base.., [8..15] at K=16+base..
// ---------------------------------------------------------------------------
template <int KTOT, bool GELU>
__global__ __launch_bounds__(256) void gemm_bf16_kernel(
    const bf16* __restrict__ A, int lda,
    const bf16* __restrict__ W, int ldb,
    const float* __restrict__ bias,
    float* __restrict__ C, int ldc) {
  const int lane = threadIdx.x & 31;
  const int wave = threadIdx.x >> 5;
  const int wm = wave >> 1;       // 0..3 -> M
  const int wn = wave & 1;        // 0..1 -> N
  const int l16 = lane & 15;
  const int half = lane >> 4;

  const int rowBase = blockIdx.y * 128 + wm * 32;
  const int colBase = blockIdx.x * 128 + wn * 64;

  const bf16* aptr[2];
  const bf16* bptr[4];
#pragma unroll
  for (int i = 0; i < 2; i++)
    aptr[i] = A + (size_t)(rowBase + i * 16 + l16) * lda + half * 8;
#pragma unroll
  for (int i = 0; i < 4; i++)
    bptr[i] = W + (size_t)(colBase + i * 16 + l16) * ldb + half * 8;

  v8f acc[2][4];
#pragma unroll
  for (int mi = 0; mi < 2; mi++)
#pragma unroll
    for (int ni = 0; ni < 4; ni++)
#pragma unroll
      for (int e = 0; e < 8; e++) acc[mi][ni][e] = 0.0f;

  for (int kk = 0; kk < KTOT; kk += 32) {
    v16bf af[2], bw[4];
#pragma unroll
    for (int i = 0; i < 2; i++) {
      v8bf alo = *(const v8bf*)(aptr[i] + kk);
      v8bf ahi = *(const v8bf*)(aptr[i] + kk + 16);
#pragma unroll
      for (int e = 0; e < 8; e++) {
        af[i][e] = alo[e];
        af[i][e + 8] = ahi[e];
      }
    }
#pragma unroll
    for (int i = 0; i < 4; i++) {
      v8bf blo = *(const v8bf*)(bptr[i] + kk);
      v8bf bhi = *(const v8bf*)(bptr[i] + kk + 16);
#pragma unroll
      for (int e = 0; e < 8; e++) {
        bw[i][e] = blo[e];
        bw[i][e + 8] = bhi[e];
      }
    }
    // Unconditional speculative prefetch of next A chunk (OOB is dropped).
    __builtin_prefetch(aptr[0] + kk + 32, 0, 0);
    __builtin_prefetch(aptr[1] + kk + 32, 0, 0);
#pragma unroll
    for (int mi = 0; mi < 2; mi++)
#pragma unroll
      for (int ni = 0; ni < 4; ni++)
        acc[mi][ni] = __builtin_amdgcn_wmma_f32_16x16x32_bf16(
            false, af[mi], false, bw[ni], (short)0, acc[mi][ni], false, false);
  }

  // Epilogue: C/D layout -> VGPR r, lanes 0-15: M=r, lanes 16-31: M=8+r; N=lane&15
#pragma unroll
  for (int mi = 0; mi < 2; mi++)
#pragma unroll
    for (int ni = 0; ni < 4; ni++) {
      const int col = colBase + ni * 16 + l16;
      const float bv = bias[col];
#pragma unroll
      for (int r = 0; r < 8; r++) {
        const int row = rowBase + mi * 16 + half * 8 + r;
        float v = acc[mi][ni][r] + bv;
        if (GELU) v = 0.5f * v * (1.0f + erff(v * 0.7071067811865475f));
        C[(size_t)row * ldc + col] = v;
      }
    }
}

// ---------------------------------------------------------------------------
// Per-row: l2norm(h) -> sim with protoN -> softmax(P=32) -> retrieved ->
// write bf16 into combined[:, 1024:2048]
// ---------------------------------------------------------------------------
__global__ __launch_bounds__(256) void softmax_retrieve_kernel(
    const float* __restrict__ h, const float* __restrict__ protoN,
    const float* __restrict__ proto, bf16* __restrict__ combined) {
  __shared__ float sh[ND];
  __shared__ float red[8];
  __shared__ float sh_sim[NP];
  __shared__ float sh_attn[NP];

  const int t = threadIdx.x;
  const size_t row = blockIdx.x;

  float ss = 0.f;
#pragma unroll
  for (int i = 0; i < 4; i++) {
    int idx = t + 256 * i;
    float v = h[row * ND + idx];
    sh[idx] = v;
    ss += v * v;
  }
  for (int m = 16; m >= 1; m >>= 1) ss += __shfl_xor(ss, m);
  if ((t & 31) == 0) red[t >> 5] = ss;
  __syncthreads();
  float tot = red[0] + red[1] + red[2] + red[3] + red[4] + red[5] + red[6] + red[7];
  const float inv = 1.0f / fmaxf(sqrtf(tot), 1e-12f);

  // wave w computes protos 4w..4w+3
  const int lane = t & 31;
  const int wave = t >> 5;
  float d0 = 0.f, d1 = 0.f, d2 = 0.f, d3 = 0.f;
  const float* p0 = protoN + (size_t)(wave * 4 + 0) * ND;
  const float* p1 = protoN + (size_t)(wave * 4 + 1) * ND;
  const float* p2 = protoN + (size_t)(wave * 4 + 2) * ND;
  const float* p3 = protoN + (size_t)(wave * 4 + 3) * ND;
  for (int k = lane; k < ND; k += 32) {
    float hv = sh[k];
    d0 += hv * p0[k];
    d1 += hv * p1[k];
    d2 += hv * p2[k];
    d3 += hv * p3[k];
  }
  for (int m = 16; m >= 1; m >>= 1) {
    d0 += __shfl_xor(d0, m);
    d1 += __shfl_xor(d1, m);
    d2 += __shfl_xor(d2, m);
    d3 += __shfl_xor(d3, m);
  }
  if (lane == 0) {
    sh_sim[wave * 4 + 0] = d0;
    sh_sim[wave * 4 + 1] = d1;
    sh_sim[wave * 4 + 2] = d2;
    sh_sim[wave * 4 + 3] = d3;
  }
  __syncthreads();

  if (t < 32) {
    float s = sh_sim[t] * inv * 10.0f;  // /TEMP, TEMP=0.1
    float mx = s;
    for (int m = 16; m >= 1; m >>= 1) mx = fmaxf(mx, __shfl_xor(mx, m));
    float e = __expf(s - mx);
    float sum = e;
    for (int m = 16; m >= 1; m >>= 1) sum += __shfl_xor(sum, m);
    sh_attn[t] = e / sum;
  }
  __syncthreads();

  // retrieved = attn @ prototypes (original input protos)
#pragma unroll
  for (int i = 0; i < 4; i++) {
    int d = t + 256 * i;
    float acc = 0.f;
#pragma unroll
    for (int p = 0; p < NP; p++) acc += sh_attn[p] * proto[(size_t)p * ND + d];
    combined[row * 2048 + ND + d] = (bf16)acc;
  }
}

// ---------------------------------------------------------------------------
// y = gelu_out + x ; LayerNorm over D; in-place on y
// ---------------------------------------------------------------------------
__global__ __launch_bounds__(256) void residual_ln_kernel(
    float* __restrict__ y, const float* __restrict__ x,
    const float* __restrict__ gamma, const float* __restrict__ beta) {
  __shared__ float red[8], red2[8];
  const int t = threadIdx.x;
  const size_t row = blockIdx.x;
  float v[4];
  float s = 0.f, s2 = 0.f;
#pragma unroll
  for (int i = 0; i < 4; i++) {
    int idx = t + 256 * i;
    float g = y[row * ND + idx];
    float xv = x[row * ND + idx];
    float yy = g + xv;
    v[i] = yy;
    s += yy;
    s2 += yy * yy;
  }
  for (int m = 16; m >= 1; m >>= 1) {
    s += __shfl_xor(s, m);
    s2 += __shfl_xor(s2, m);
  }
  if ((t & 31) == 0) {
    red[t >> 5] = s;
    red2[t >> 5] = s2;
  }
  __syncthreads();
  float ts = red[0] + red[1] + red[2] + red[3] + red[4] + red[5] + red[6] + red[7];
  float ts2 = red2[0] + red2[1] + red2[2] + red2[3] + red2[4] + red2[5] + red2[6] + red2[7];
  const float mu = ts * (1.0f / ND);
  const float var = ts2 * (1.0f / ND) - mu * mu;
  const float rstd = rsqrtf(var + 1e-5f);
#pragma unroll
  for (int i = 0; i < 4; i++) {
    int idx = t + 256 * i;
    y[row * ND + idx] = (v[i] - mu) * rstd * gamma[idx] + beta[idx];
  }
}

// ---------------------------------------------------------------------------
extern "C" void kernel_launch(void* const* d_in, const int* in_sizes, int n_in,
                              void* d_out, int out_size, void* d_ws, size_t ws_size,
                              hipStream_t stream) {
  const float* x     = (const float*)d_in[0];
  const float* Wi    = (const float*)d_in[1];
  const float* bi    = (const float*)d_in[2];
  const float* Wo    = (const float*)d_in[3];
  const float* bo    = (const float*)d_in[4];
  const float* gamma = (const float*)d_in[5];
  const float* beta  = (const float*)d_in[6];
  const float* proto = (const float*)d_in[7];
  float* out = (float*)d_out;   // reused as h, then gelu-out, then final y

  char* ws = (char*)d_ws;
  bf16* combined = (bf16*)ws;                                   // B*2D bf16 = 256 MB
  bf16* WiB      = (bf16*)(ws + (size_t)NB * 2 * ND * sizeof(bf16));
  bf16* WoB      = WiB + (size_t)ND * ND;                       // Wi: 2 MB
  float* protoN  = (float*)(WoB + (size_t)ND * 2 * ND);         // Wo: 4 MB

  // Downconvert weights + x, normalize prototypes
  f32_to_bf16_kernel<<<(ND * ND) / 256, 256, 0, stream>>>(Wi, WiB, ND * ND);
  f32_to_bf16_kernel<<<(ND * 2 * ND) / 256, 256, 0, stream>>>(Wo, WoB, ND * 2 * ND);
  pack_x_kernel<<<(int)(((size_t)NB * ND) / 256), 256, 0, stream>>>(x, combined);
  proto_norm_kernel<<<NP, 256, 0, stream>>>(proto, protoN);

  dim3 grid(ND / 128, NB / 128);   // N-major for L2 reuse of A tiles

  // h = x @ Wi^T + bi    (A = combined[:, :1024], lda = 2048)
  gemm_bf16_kernel<1024, false><<<grid, 256, 0, stream>>>(
      combined, 2048, WiB, ND, bi, out, ND);

  // softmax over prototypes; write retrieved bf16 into combined[:, 1024:]
  softmax_retrieve_kernel<<<NB, 256, 0, stream>>>(out, protoN, proto, combined);

  // out = gelu(combined @ Wo^T + bo)
  gemm_bf16_kernel<2048, true><<<grid, 256, 0, stream>>>(
      combined, 2048, WoB, 2 * ND, bo, out, ND);

  // y = LN(out + x) * gamma + beta
  residual_ln_kernel<<<NB, 256, 0, stream>>>(out, x, gamma, beta);
}
